// QuantumToroidalCore_58480274702987
// MI455X (gfx1250) — matrix operand order repaired and checked by reading, not compiled, not run on hardware
//
#include <hip/hip_runtime.h>

// ---------------- problem constants ----------------
#define SDIM 1024
#define BDIM 4
#define DDIM 1024
#define HDIM 16
#define DHD  64
#define LNUM 27
#define MROWS (SDIM*BDIM)   // 4096 rows of the flattened [S,B] activation matrix

typedef unsigned short u16;
typedef unsigned int   u32;

typedef __attribute__((ext_vector_type(16))) __bf16 v16bf;
typedef __attribute__((ext_vector_type(8)))  float  v8f;

union FragAB { uint4 q[2]; v16bf v; };

__device__ __forceinline__ u16 f2b(float f) {
    u32 u = __float_as_uint(f);
    u32 r = u + 0x7FFFu + ((u >> 16) & 1u);   // round-to-nearest-even
    return (u16)(r >> 16);
}
__device__ __forceinline__ float b2f(u16 h) {
    return __uint_as_float(((u32)h) << 16);
}

// async copy of 16B global -> LDS (per lane), tracked with ASYNCcnt.
// INST_OFFSET is added to BOTH the LDS and global addresses (ISA 10.x async pseudocode),
// so one base pair covers a 32B row chunk with offsets 0 and 16.
__device__ __forceinline__ void async_b128(u32 lds_addr, const void* gaddr) {
    asm volatile("global_load_async_to_lds_b128 %0, %1, off"
                 :: "v"(lds_addr), "v"(gaddr) : "memory");
}
__device__ __forceinline__ void async_b128_off16(u32 lds_addr, const void* gaddr) {
    asm volatile("global_load_async_to_lds_b128 %0, %1, off offset:16"
                 :: "v"(lds_addr), "v"(gaddr) : "memory");
}
__device__ __forceinline__ void wait_async0() {
    asm volatile("s_wait_asynccnt 0x0" ::: "memory");
}
__device__ __forceinline__ u32 lds_addr_of(const void* p) {
    return (u32)(size_t)p;   // LDS aperture keeps byte offset in addr[31:0]
}

// ---------------- fp32 -> bf16 convert ----------------
__global__ void k_f32_to_bf16(const float* __restrict__ in, u16* __restrict__ out, int n) {
    int i = blockIdx.x * blockDim.x + threadIdx.x;
    int stride = gridDim.x * blockDim.x;
    for (; i < n; i += stride) out[i] = f2b(in[i]);
}

// ---------------- WMMA GEMM: C[M,N] = A[M,K](bf16) @ Wb[N,K]^T(bf16) + bias ----------------
// Double-buffered LDS, async global->LDS staging for both operands.
// epilogue: optional residual (bf16), optional stage modulation, bf16 out (+fp32 optional).
__launch_bounds__(256)
__global__ void k_gemm_wmma(const u16*  __restrict__ A,
                            const u16*  __restrict__ Wb,
                            const float* __restrict__ bias,
                            u16*   __restrict__ Cb,
                            float* __restrict__ Cf,
                            const u16* __restrict__ Res,
                            const int* __restrict__ stagep,
                            int M, int N, int K, int apply_stage)
{
    __shared__ __align__(16) u16 As[2][128 * 40];   // 128 rows x 32 k, pitch 40
    __shared__ __align__(16) u16 Bs[2][128 * 40];   // 128 cols x 32 k, pitch 40

    const int tid  = threadIdx.x;
    const int lane = tid & 31;
    const int wid  = tid >> 5;          // 0..7
    const int hi   = lane >> 4;         // lane half
    const int ln   = lane & 15;

    const int m0 = (wid & 3) * 32;      // wave row offset within 128-tile
    const int n0 = (wid >> 2) * 64;     // wave col offset within 128-tile
    const int blockM = blockIdx.y * 128;
    const int blockN = blockIdx.x * 128;

    v8f acc[2][4];
    #pragma unroll
    for (int i = 0; i < 2; i++)
        #pragma unroll
        for (int j = 0; j < 4; j++)
            #pragma unroll
            for (int r = 0; r < 8; r++) acc[i][j][r] = 0.0f;

    const int srow = tid >> 1;          // 0..127 (staging row)
    const int scol = (tid & 1) * 16;    // 0 or 16 (staging k offset)

    const u16* agBase = A  + (size_t)(blockM + srow) * K + scol;
    const u16* bgBase = Wb + (size_t)(blockN + srow) * K + scol;
    const u32 ldsA[2] = { lds_addr_of(&As[0][srow * 40 + scol]),
                          lds_addr_of(&As[1][srow * 40 + scol]) };
    const u32 ldsB[2] = { lds_addr_of(&Bs[0][srow * 40 + scol]),
                          lds_addr_of(&Bs[1][srow * 40 + scol]) };

    // prologue: stage buffer 0
    {
        async_b128      (ldsA[0], agBase);
        async_b128_off16(ldsA[0], agBase);
        async_b128      (ldsB[0], bgBase);
        async_b128_off16(ldsB[0], bgBase);
    }

    int buf = 0;
    for (int kt = 0; kt < K; kt += 32) {
        wait_async0();        // own async copies into buf complete
        __syncthreads();      // everyone's copies complete -> buf valid for all waves

        // stage the other buffer for the next iteration (overlaps with compute)
        if (kt + 32 < K) {
            int nb = buf ^ 1;
            const u16* ag = agBase + (kt + 32);
            const u16* bg = bgBase + (kt + 32);
            async_b128      (ldsA[nb], ag);
            async_b128_off16(ldsA[nb], ag);
            async_b128      (ldsB[nb], bg);
            async_b128_off16(ldsB[nb], bg);
        }

        // fragments (ISA wave32 layouts)
        FragAB a[2], b[4];
        #pragma unroll
        for (int mt = 0; mt < 2; mt++) {
            int r = m0 + mt * 16 + ln;
            a[mt].q[0] = *(const uint4*)&As[buf][r * 40 + 8 * hi];        // K = 8*hi..+7
            a[mt].q[1] = *(const uint4*)&As[buf][r * 40 + 16 + 8 * hi];   // K = 16+8*hi..+7
        }
        #pragma unroll
        for (int nt = 0; nt < 4; nt++) {
            int c = n0 + nt * 16 + ln;
            b[nt].q[0] = *(const uint4*)&Bs[buf][c * 40 + 8 * hi];        // K = 8*hi..+7
            b[nt].q[1] = *(const uint4*)&Bs[buf][c * 40 + 16 + 8 * hi];   // K = 16+8*hi..+7
        }
        #pragma unroll
        for (int mt = 0; mt < 2; mt++)
            #pragma unroll
            for (int nt = 0; nt < 4; nt++)
                acc[mt][nt] = __builtin_amdgcn_wmma_f32_16x16x32_bf16(
                    false, a[mt].v, false, b[nt].v, (short)0, acc[mt][nt], false, false);
        buf ^= 1;
    }

    float gain = 1.0f; int use_tanh = 0;
    if (apply_stage) {
        int st = stagep ? *stagep : 5;
        gain = (st == 0) ? 0.7f : ((st == 5) ? 1.3f : ((st == 9) ? 2.0f : 1.0f));
        use_tanh = (st == 8);
    }

    #pragma unroll
    for (int nt = 0; nt < 4; nt++) {
        int col = blockN + n0 + nt * 16 + ln;
        float bv = bias[col];
        #pragma unroll
        for (int mt = 0; mt < 2; mt++) {
            #pragma unroll
            for (int r = 0; r < 8; r++) {
                int row = blockM + m0 + mt * 16 + r + 8 * hi;
                size_t idx = (size_t)row * N + col;
                float v = acc[mt][nt][r] + bv;
                if (Res) v += b2f(Res[idx]);
                if (apply_stage) v = use_tanh ? tanhf(v) : v * gain;
                Cb[idx] = f2b(v);
                if (Cf) Cf[idx] = v;
            }
        }
    }
}

// ---------------- flash attention with toroidal band bias ----------------
// qkv: [MROWS, 3*D] bf16, row = s*B + b.  out: [MROWS, D] bf16.
// block = one (b,h) x 128 query rows; wave owns 16 query rows.
__launch_bounds__(256)
__global__ void k_attention(const u16* __restrict__ qkv, u16* __restrict__ out)
{
    __shared__ __align__(16) u16 Ks[32 * 72];     // [key][dh], pitch 72 (async-staged)
    __shared__ __align__(16) u16 Vt[64 * 40];     // [dh][key], pitch 40 (transposed)
    __shared__ __align__(16) u16 Ps[8][16 * 40];  // per-wave P tile [row][key], pitch 40

    const int tid  = threadIdx.x;
    const int lane = tid & 31;
    const int wid  = tid >> 5;
    const int hi   = lane >> 4;
    const int ln   = lane & 15;

    const int bh = blockIdx.y;
    const int bb = bh / HDIM;
    const int hh = bh % HDIM;
    const int q0 = blockIdx.x * 128 + wid * 16;

    const int rowpitch = 3 * DDIM;

    // Q fragments: A-operand 16x64 -> two K-steps of 32 (direct from global)
    FragAB aq[2];
    {
        const u16* qbase = qkv + ((size_t)(q0 + ln) * BDIM + bb) * rowpitch + hh * DHD;
        #pragma unroll
        for (int ks = 0; ks < 2; ks++) {
            aq[ks].q[0] = *(const uint4*)(qbase + ks * 32 + 8 * hi);
            aq[ks].q[1] = *(const uint4*)(qbase + ks * 32 + 16 + 8 * hi);
        }
    }

    v8f o[4];
    #pragma unroll
    for (int t = 0; t < 4; t++)
        #pragma unroll
        for (int r = 0; r < 8; r++) o[t][r] = 0.0f;
    float m_row[8], l_row[8];
    #pragma unroll
    for (int r = 0; r < 8; r++) { m_row[r] = -1e30f; l_row[r] = 0.0f; }

    const int keyIdx = tid >> 3;        // 0..31
    const int dh0    = (tid & 7) * 8;   // 0..56
    const u32 ksAddr = lds_addr_of(&Ks[keyIdx * 72 + dh0]);

    for (int k0 = 0; k0 < SDIM; k0 += 32) {
        // stage K (async global->LDS) and V (VGPR transpose) tiles
        {
            const u16* kg = qkv + ((size_t)(k0 + keyIdx) * BDIM + bb) * rowpitch
                            + DDIM + hh * DHD + dh0;
            async_b128(ksAddr, kg);
            uint4 vq = *(const uint4*)(kg + DDIM);
            const u16* vs = (const u16*)&vq;
            #pragma unroll
            for (int j = 0; j < 8; j++) Vt[(dh0 + j) * 40 + keyIdx] = vs[j];
        }
        wait_async0();
        __syncthreads();

        // scores: two 16x16 tiles over keys [k0..+15], [k0+16..+31]
        v8f s[2];
        #pragma unroll
        for (int t = 0; t < 2; t++)
            #pragma unroll
            for (int r = 0; r < 8; r++) s[t][r] = 0.0f;
        #pragma unroll
        for (int t = 0; t < 2; t++) {
            int c = t * 16 + ln;  // local key (B-operand N index)
            #pragma unroll
            for (int ks = 0; ks < 2; ks++) {
                FragAB bk;
                bk.q[0] = *(const uint4*)&Ks[c * 72 + ks * 32 + 8 * hi];
                bk.q[1] = *(const uint4*)&Ks[c * 72 + ks * 32 + 16 + 8 * hi];
                s[t] = __builtin_amdgcn_wmma_f32_16x16x32_bf16(
                    false, aq[ks].v, false, bk.v, (short)0, s[t], false, false);
            }
        }

        // bias + online softmax
        #pragma unroll
        for (int r = 0; r < 8; r++) {
            int i  = q0 + r + 8 * hi;
            int j0 = k0 + ln;
            float sc0 = s[0][r] * 0.125f;
            u32 d0 = (u32)(i - j0 + SDIM) & (SDIM - 1);
            if (d0 <= 1u || d0 == (SDIM - 1)) sc0 += 1.0f;
            float sc1 = s[1][r] * 0.125f;
            u32 d1 = (u32)(i - j0 - 16 + SDIM) & (SDIM - 1);
            if (d1 <= 1u || d1 == (SDIM - 1)) sc1 += 1.0f;

            float mx = fmaxf(sc0, sc1);
            mx = fmaxf(mx, __shfl_xor(mx, 1));
            mx = fmaxf(mx, __shfl_xor(mx, 2));
            mx = fmaxf(mx, __shfl_xor(mx, 4));
            mx = fmaxf(mx, __shfl_xor(mx, 8));
            float m_new = fmaxf(m_row[r], mx);
            float alpha = __expf(m_row[r] - m_new);
            float p0 = __expf(sc0 - m_new);
            float p1 = __expf(sc1 - m_new);
            float ps = p0 + p1;
            ps += __shfl_xor(ps, 1);
            ps += __shfl_xor(ps, 2);
            ps += __shfl_xor(ps, 4);
            ps += __shfl_xor(ps, 8);
            l_row[r] = l_row[r] * alpha + ps;
            m_row[r] = m_new;
            #pragma unroll
            for (int t = 0; t < 4; t++) o[t][r] *= alpha;

            int prow = r + 8 * hi;
            Ps[wid][prow * 40 + ln]      = f2b(p0);
            Ps[wid][prow * 40 + ln + 16] = f2b(p1);
        }

        // P (16x32) @ V (32x64): re-layout P through LDS into an A fragment
        FragAB ap;
        ap.q[0] = *(const uint4*)&Ps[wid][ln * 40 + 8 * hi];
        ap.q[1] = *(const uint4*)&Ps[wid][ln * 40 + 16 + 8 * hi];
        #pragma unroll
        for (int t = 0; t < 4; t++) {
            int dh = t * 16 + ln;
            FragAB bv;
            bv.q[0] = *(const uint4*)&Vt[dh * 40 + 8 * hi];
            bv.q[1] = *(const uint4*)&Vt[dh * 40 + 16 + 8 * hi];
            o[t] = __builtin_amdgcn_wmma_f32_16x16x32_bf16(
                false, ap.v, false, bv.v, (short)0, o[t], false, false);
        }
        __syncthreads();
    }

    // normalize + store
    #pragma unroll
    for (int t = 0; t < 4; t++) {
        #pragma unroll
        for (int r = 0; r < 8; r++) {
            int sq = q0 + r + 8 * hi;
            float val = o[t][r] / l_row[r];
            out[((size_t)sq * BDIM + bb) * DDIM + hh * DHD + t * 16 + ln] = f2b(val);
        }
    }
}

// ---------------- score heads ----------------
__global__ void k_zero2(float* a) { if (threadIdx.x == 0) { a[0] = 0.0f; a[1] = 0.0f; } }

__global__ void k_reduce(const float* __restrict__ x, const float* __restrict__ mw,
                         const float* __restrict__ yw, float* __restrict__ acc)
{
    __shared__ float r0[256], r1[256];
    float s0 = 0.0f, s1 = 0.0f;
    const int n = MROWS * DDIM;
    for (int i = blockIdx.x * blockDim.x + threadIdx.x; i < n; i += gridDim.x * blockDim.x) {
        int col = i & (DDIM - 1);
        float v = x[i];
        s0 += v * mw[col];
        s1 += v * yw[col];
    }
    r0[threadIdx.x] = s0; r1[threadIdx.x] = s1;
    __syncthreads();
    for (int s = 128; s > 0; s >>= 1) {
        if (threadIdx.x < (unsigned)s) {
            r0[threadIdx.x] += r0[threadIdx.x + s];
            r1[threadIdx.x] += r1[threadIdx.x + s];
        }
        __syncthreads();
    }
    if (threadIdx.x == 0) { atomicAdd(&acc[0], r0[0]); atomicAdd(&acc[1], r1[0]); }
}

__global__ void k_finalize(const float* __restrict__ acc, float* __restrict__ out2) {
    if (threadIdx.x == 0) {
        out2[0] = 1.0f / (1.0f + __expf(-acc[0] / (float)MROWS));
        out2[1] = 1.0f / (1.0f + __expf(-acc[1] / (float)MROWS));
    }
}

// ---------------- host orchestration ----------------
static inline void launch_gemm(const u16* A, const u16* Wb, const float* bias,
                               u16* Cb, float* Cf, const u16* Res, const int* stage,
                               int M, int N, int K, int apply_stage, hipStream_t s)
{
    dim3 grid(N / 128, M / 128), block(256);
    k_gemm_wmma<<<grid, block, 0, s>>>(A, Wb, bias, Cb, Cf, Res, stage, M, N, K, apply_stage);
}

extern "C" void kernel_launch(void* const* d_in, const int* in_sizes, int n_in,
                              void* d_out, int out_size, void* d_ws, size_t ws_size,
                              hipStream_t stream)
{
    const float* x     = (const float*)d_in[0];
    const float* qe_w  = (const float*)d_in[1];
    const float* qe_b  = (const float*)d_in[2];
    const float* tp_w  = (const float*)d_in[3];
    const float* tp_b  = (const float*)d_in[4];
    const float* in_w  = (const float*)d_in[5];
    const float* in_b  = (const float*)d_in[6];
    const float* out_w = (const float*)d_in[7];
    const float* out_b = (const float*)d_in[8];
    const float* maat  = (const float*)d_in[9];
    const float* yunus = (const float*)d_in[10];
    const int*   stage = (const int*)d_in[11];

    // workspace layout (bf16 activation buffers + per-layer weight staging)
    char* ws = (char*)d_ws;
    size_t off = 0;
    u16* hA   = (u16*)(ws + off); off += (size_t)MROWS * DDIM * 2;        // 8 MB
    u16* hB   = (u16*)(ws + off); off += (size_t)MROWS * DDIM * 2;        // 8 MB
    u16* xtb  = (u16*)(ws + off); off += (size_t)MROWS * DDIM * 2;        // 8 MB (also attn out)
    u16* qkvb = (u16*)(ws + off); off += (size_t)MROWS * 3 * DDIM * 2;    // 24 MB
    u16* wbuf = (u16*)(ws + off); off += (size_t)3 * DDIM * DDIM * 2;     // 6 MB weight staging
    float* accs = (float*)(ws + off);                                     // 2 floats

    float* outx = (float*)d_out;

    // x -> bf16 (into xtb as temp), then quantum embedding into hA
    k_f32_to_bf16<<<1024, 256, 0, stream>>>(x, xtb, MROWS * DDIM);
    k_f32_to_bf16<<<1024, 256, 0, stream>>>(qe_w, wbuf, DDIM * DDIM);
    launch_gemm(xtb, wbuf, qe_b, hA, nullptr, nullptr, nullptr,
                MROWS, DDIM, DDIM, 0, stream);

    u16* cur = hA;
    u16* nxt = hB;
    for (int l = 0; l < LNUM; l++) {
        const float* tw = tp_w  + (size_t)l * DDIM * DDIM;
        const float* tb = tp_b  + (size_t)l * DDIM;
        const float* iw = in_w  + (size_t)l * 3 * DDIM * DDIM;
        const float* ib = in_b  + (size_t)l * 3 * DDIM;
        const float* owp = out_w + (size_t)l * DDIM * DDIM;
        const float* obp = out_b + (size_t)l * DDIM;

        // toroidal projection
        k_f32_to_bf16<<<1024, 256, 0, stream>>>(tw, wbuf, DDIM * DDIM);
        launch_gemm(cur, wbuf, tb, xtb, nullptr, nullptr, nullptr,
                    MROWS, DDIM, DDIM, 0, stream);
        // qkv projection
        k_f32_to_bf16<<<1024, 256, 0, stream>>>(iw, wbuf, 3 * DDIM * DDIM);
        launch_gemm(xtb, wbuf, ib, qkvb, nullptr, nullptr, nullptr,
                    MROWS, 3 * DDIM, DDIM, 0, stream);
        // attention (writes into xtb, which is free now)
        {
            dim3 grid(SDIM / 128, BDIM * HDIM), block(256);
            k_attention<<<grid, block, 0, stream>>>(qkvb, xtb);
        }
        // out projection + fractal residual + stage modulation
        int f = ((l % 3) == 2);
        int last = (l == LNUM - 1);
        k_f32_to_bf16<<<1024, 256, 0, stream>>>(owp, wbuf, DDIM * DDIM);
        launch_gemm(xtb, wbuf, obp, nxt, last ? outx : nullptr,
                    f ? cur : nullptr, stage, MROWS, DDIM, DDIM, 1, stream);

        u16* t = cur; cur = nxt; nxt = t;
    }

    // truth / safety scores
    k_zero2<<<1, 32, 0, stream>>>(accs);
    k_reduce<<<256, 256, 0, stream>>>(outx, maat, yunus, accs);
    k_finalize<<<1, 32, 0, stream>>>(accs, outx + (size_t)MROWS * DDIM);
}
